// OptimizedGroupedExperts_18451179504175
// MI455X (gfx1250) — matrix operand order repaired and checked by reading, not compiled, not run on hardware
//
#include <hip/hip_runtime.h>

// ---------------- problem constants ----------------
#define N_TOKENS  8192
#define D_MODEL   1024
#define D_FF      4096
#define TOP_K     2
#define N_EXPERTS 8
#define TOTAL     (N_TOKENS * TOP_K)          // 16384
#define CAP       4096                         // int(2.0 * TOTAL / N_EXPERTS)

#define KSTEP     32
#define LDSTRIDE  40                           // 32 K-elements + 8 pad (80B rows: conflict-free b128 reads)

typedef __bf16 bf16;
typedef __attribute__((ext_vector_type(16))) __bf16 v16bf;
typedef __attribute__((ext_vector_type(4)))  __bf16 v4bf;
typedef __attribute__((ext_vector_type(8)))  float  v8f;

union Frag16 { v16bf v; uint4 u[2]; };

// ---------------- CDNA5 async global->LDS helpers ----------------
// clang diagnostic revealed: param0 = v4i addrspace(1)* (global src), 4 args total.
typedef int v4i __attribute__((vector_size(16)));
typedef __attribute__((address_space(1))) v4i gv4i;
typedef __attribute__((address_space(3))) v4i lv4i;

__device__ __forceinline__ void cp_async_b128(bf16* lds_dst, const bf16* gsrc) {
#if __has_builtin(__builtin_amdgcn_global_load_async_to_lds_b128)
    __builtin_amdgcn_global_load_async_to_lds_b128((gv4i*)gsrc, (lv4i*)lds_dst, 0, 0);
#else
    *(uint4*)lds_dst = *(const uint4*)gsrc;    // synchronous fallback
#endif
}

__device__ __forceinline__ void async_wait0() {
#if __has_builtin(__builtin_amdgcn_s_wait_asynccnt)
    __builtin_amdgcn_s_wait_asynccnt(0);
#else
    asm volatile("s_wait_asynccnt 0x0" ::: "memory");
#endif
}

// ---------------- init: zero output + counters ----------------
__global__ void k_init(float* __restrict__ out, int n4, int* __restrict__ counts) {
    int i = blockIdx.x * blockDim.x + threadIdx.x;
    if (i < n4) {
        float4 z; z.x = 0.f; z.y = 0.f; z.z = 0.f; z.w = 0.f;
        ((float4*)out)[i] = z;
    }
    if (i < N_EXPERTS) counts[i] = 0;
}

// ---------------- x: fp32 -> bf16 ----------------
__global__ void k_cvt_x(const float* __restrict__ in, bf16* __restrict__ out, int n4) {
    int i = blockIdx.x * blockDim.x + threadIdx.x;
    if (i >= n4) return;
    float4 f = ((const float4*)in)[i];
    v4bf o = { (bf16)f.x, (bf16)f.y, (bf16)f.z, (bf16)f.w };
    ((v4bf*)out)[i] = o;
}

// ---------------- weight transpose + convert: [R][C] fp32 -> [C][R] bf16 ----------------
__global__ void k_transpose_bf16(const float* __restrict__ in, bf16* __restrict__ out,
                                 int R, int C) {
    __shared__ float tile[32][33];
    int e = blockIdx.z;
    const float* src = in  + (size_t)e * R * C;
    bf16*        dst = out + (size_t)e * R * C;
    int c0 = blockIdx.x * 32, r0 = blockIdx.y * 32;
    int tx = threadIdx.x, ty = threadIdx.y;     // block = (32, 8)
#pragma unroll
    for (int i = 0; i < 32; i += 8)
        tile[ty + i][tx] = src[(size_t)(r0 + ty + i) * C + (c0 + tx)];
    __syncthreads();
#pragma unroll
    for (int i = 0; i < 32; i += 8)
        dst[(size_t)(c0 + ty + i) * R + (r0 + tx)] = (bf16)tile[tx][ty + i];
}

// ---------------- routing ----------------
__global__ void k_route_count(const int* __restrict__ idx, int* __restrict__ counts) {
    int i = blockIdx.x * blockDim.x + threadIdx.x;
    if (i < TOTAL) atomicAdd(&counts[idx[i]], 1);
}

__global__ void k_prefix(const int* __restrict__ counts, int* __restrict__ offsets,
                         int* __restrict__ counters, int* __restrict__ cnts) {
    int off = 0;
    for (int e = 0; e < N_EXPERTS; ++e) {
        int c = counts[e]; if (c > CAP) c = CAP;
        offsets[e] = off; cnts[e] = c; counters[e] = 0;
        off += c;
    }
    offsets[N_EXPERTS] = off;
}

__global__ void k_route_assign(const int* __restrict__ idx, const float* __restrict__ wts,
                               int* __restrict__ counters, const int* __restrict__ offsets,
                               const int* __restrict__ cnts,
                               int* __restrict__ row_token, float* __restrict__ row_weight) {
    int i = blockIdx.x * blockDim.x + threadIdx.x;
    if (i >= TOTAL) return;
    int e = idx[i];
    int pos = atomicAdd(&counters[e], 1);
    if (pos < cnts[e]) {
        int g = offsets[e] + pos;
        row_token[g]  = i / TOP_K;
        row_weight[g] = wts[i];
    }
}

// ---------------- GEMM1: gate/value + SwiGLU, bf16 WMMA, async-LDS double buffer ----------------
// 512 threads = 16 waves (4x4), block tile 128(M) x 128(N), wave tile 32x32, dual B (W1,W2)
__global__ __launch_bounds__(512) void k_ffn1(
    const bf16* __restrict__ xb, const bf16* __restrict__ w1t,
    const bf16* __restrict__ w2t, const int* __restrict__ row_token,
    const int* __restrict__ offsets, const int* __restrict__ cnts,
    bf16* __restrict__ H)
{
    __shared__ bf16 sA [2][128 * LDSTRIDE];
    __shared__ bf16 sB1[2][128 * LDSTRIDE];
    __shared__ bf16 sB2[2][128 * LDSTRIDE];

    const int e    = blockIdx.z;
    const int cnt  = cnts[e];
    const int bm   = blockIdx.y * 128;
    if (bm >= cnt) return;
    const int bn   = blockIdx.x * 128;
    const int tid  = threadIdx.x;
    const int wave = tid >> 5;
    const int lane = tid & 31;
    const int wm   = (wave & 3) * 32;
    const int wn   = (wave >> 2) * 32;
    const int half = lane >> 4;
    const int r    = lane & 15;
    const int kgrp = half * 16;
    const int goff = offsets[e];

    // staging role: thread -> one 16B chunk per matrix per k-step (128 rows x 4 chunks)
    const int srow = tid >> 2;           // 0..127
    const int skc  = (tid & 3) * 8;      // element offset within k-tile
    int mrow = bm + srow; if (mrow >= cnt) mrow = cnt - 1;
    const bf16* aSrc  = xb  + (size_t)row_token[goff + mrow] * D_MODEL + skc;
    const bf16* b1Src = w1t + ((size_t)e * D_FF + bn + srow) * D_MODEL + skc;
    const bf16* b2Src = w2t + ((size_t)e * D_FF + bn + srow) * D_MODEL + skc;
    const int sdst = srow * LDSTRIDE + skc;

    cp_async_b128(&sA [0][sdst], aSrc);
    cp_async_b128(&sB1[0][sdst], b1Src);
    cp_async_b128(&sB2[0][sdst], b2Src);

    v8f ag[2][2], av[2][2];
#pragma unroll
    for (int mi = 0; mi < 2; ++mi)
#pragma unroll
        for (int ni = 0; ni < 2; ++ni) {
            ag[mi][ni] = (v8f){0.f,0.f,0.f,0.f,0.f,0.f,0.f,0.f};
            av[mi][ni] = (v8f){0.f,0.f,0.f,0.f,0.f,0.f,0.f,0.f};
        }

    int buf = 0;
    for (int k = 0; k < D_MODEL; k += KSTEP, buf ^= 1) {
        async_wait0();                       // this wave's copies for buf are done
        __syncthreads();                     // all waves' copies visible; prev reads of buf^1 done
        if (k + KSTEP < D_MODEL) {           // prefetch next k-tile into other buffer
            int nb = buf ^ 1;
            cp_async_b128(&sA [nb][sdst], aSrc  + k + KSTEP);
            cp_async_b128(&sB1[nb][sdst], b1Src + k + KSTEP);
            cp_async_b128(&sB2[nb][sdst], b2Src + k + KSTEP);
        }

        Frag16 a[2], f1[2], f2[2];
#pragma unroll
        for (int mi = 0; mi < 2; ++mi) {
            const bf16* p = &sA[buf][(wm + mi * 16 + r) * LDSTRIDE + kgrp];
            a[mi].u[0] = *(const uint4*)(p);
            a[mi].u[1] = *(const uint4*)(p + 8);
        }
#pragma unroll
        for (int ni = 0; ni < 2; ++ni) {
            const bf16* p1 = &sB1[buf][(wn + ni * 16 + r) * LDSTRIDE + kgrp];
            f1[ni].u[0] = *(const uint4*)(p1);
            f1[ni].u[1] = *(const uint4*)(p1 + 8);
            const bf16* p2 = &sB2[buf][(wn + ni * 16 + r) * LDSTRIDE + kgrp];
            f2[ni].u[0] = *(const uint4*)(p2);
            f2[ni].u[1] = *(const uint4*)(p2 + 8);
        }
#pragma unroll
        for (int mi = 0; mi < 2; ++mi)
#pragma unroll
            for (int ni = 0; ni < 2; ++ni) {
                ag[mi][ni] = __builtin_amdgcn_wmma_f32_16x16x32_bf16(
                    false, a[mi].v, false, f1[ni].v, (short)0, ag[mi][ni], false, false);
                av[mi][ni] = __builtin_amdgcn_wmma_f32_16x16x32_bf16(
                    false, a[mi].v, false, f2[ni].v, (short)0, av[mi][ni], false, false);
            }
    }

    // SwiGLU epilogue (C layout: VGPR t -> M = t + half*8, lane&15 -> N)
#pragma unroll
    for (int mi = 0; mi < 2; ++mi)
#pragma unroll
        for (int ni = 0; ni < 2; ++ni)
#pragma unroll
            for (int t = 0; t < 8; ++t) {
                int m = bm + wm + mi * 16 + half * 8 + t;
                if (m < cnt) {
                    int n = bn + wn + ni * 16 + r;
                    float g = ag[mi][ni][t];
                    float v = av[mi][ni][t];
                    float h = (g / (1.0f + __expf(-g))) * v;
                    H[(size_t)(goff + m) * D_FF + n] = (bf16)h;
                }
            }
}

// ---------------- GEMM2: H @ W3^T + weighted scatter-add, async-LDS double buffer ----------------
__global__ __launch_bounds__(512) void k_ffn2(
    const bf16* __restrict__ H, const bf16* __restrict__ w3t,
    const int* __restrict__ row_token, const float* __restrict__ row_weight,
    const int* __restrict__ offsets, const int* __restrict__ cnts,
    float* __restrict__ out)
{
    __shared__ bf16 sA[2][128 * LDSTRIDE];
    __shared__ bf16 sB[2][128 * LDSTRIDE];

    const int e    = blockIdx.z;
    const int cnt  = cnts[e];
    const int bm   = blockIdx.y * 128;
    if (bm >= cnt) return;
    const int bn   = blockIdx.x * 128;
    const int tid  = threadIdx.x;
    const int wave = tid >> 5;
    const int lane = tid & 31;
    const int wm   = (wave & 3) * 32;
    const int wn   = (wave >> 2) * 32;
    const int half = lane >> 4;
    const int r    = lane & 15;
    const int kgrp = half * 16;
    const int goff = offsets[e];

    const int srow = tid >> 2;
    const int skc  = (tid & 3) * 8;
    int mrow = bm + srow; if (mrow >= cnt) mrow = cnt - 1;
    const bf16* aSrc = H   + (size_t)(goff + mrow) * D_FF + skc;
    const bf16* bSrc = w3t + ((size_t)e * D_MODEL + bn + srow) * D_FF + skc;
    const int sdst = srow * LDSTRIDE + skc;

    cp_async_b128(&sA[0][sdst], aSrc);
    cp_async_b128(&sB[0][sdst], bSrc);

    v8f acc[2][2];
#pragma unroll
    for (int mi = 0; mi < 2; ++mi)
#pragma unroll
        for (int ni = 0; ni < 2; ++ni)
            acc[mi][ni] = (v8f){0.f,0.f,0.f,0.f,0.f,0.f,0.f,0.f};

    int buf = 0;
    for (int k = 0; k < D_FF; k += KSTEP, buf ^= 1) {
        async_wait0();
        __syncthreads();
        if (k + KSTEP < D_FF) {
            int nb = buf ^ 1;
            cp_async_b128(&sA[nb][sdst], aSrc + k + KSTEP);
            cp_async_b128(&sB[nb][sdst], bSrc + k + KSTEP);
        }

        Frag16 a[2], fb[2];
#pragma unroll
        for (int mi = 0; mi < 2; ++mi) {
            const bf16* p = &sA[buf][(wm + mi * 16 + r) * LDSTRIDE + kgrp];
            a[mi].u[0] = *(const uint4*)(p);
            a[mi].u[1] = *(const uint4*)(p + 8);
        }
#pragma unroll
        for (int ni = 0; ni < 2; ++ni) {
            const bf16* p = &sB[buf][(wn + ni * 16 + r) * LDSTRIDE + kgrp];
            fb[ni].u[0] = *(const uint4*)(p);
            fb[ni].u[1] = *(const uint4*)(p + 8);
        }
#pragma unroll
        for (int mi = 0; mi < 2; ++mi)
#pragma unroll
            for (int ni = 0; ni < 2; ++ni)
                acc[mi][ni] = __builtin_amdgcn_wmma_f32_16x16x32_bf16(
                    false, a[mi].v, false, fb[ni].v, (short)0, acc[mi][ni], false, false);
    }

#pragma unroll
    for (int mi = 0; mi < 2; ++mi)
#pragma unroll
        for (int ni = 0; ni < 2; ++ni)
#pragma unroll
            for (int t = 0; t < 8; ++t) {
                int m = bm + wm + mi * 16 + half * 8 + t;
                if (m < cnt) {
                    int g   = goff + m;
                    int tok = row_token[g];
                    float w = row_weight[g];
                    int n   = bn + wn + ni * 16 + r;
                    atomicAdd(&out[(size_t)tok * D_MODEL + n], acc[mi][ni][t] * w);
                }
            }
}

// ---------------- host-side launch ----------------
extern "C" void kernel_launch(void* const* d_in, const int* in_sizes, int n_in,
                              void* d_out, int out_size, void* d_ws, size_t ws_size,
                              hipStream_t stream) {
    (void)in_sizes; (void)n_in; (void)out_size; (void)ws_size;
    const float* x   = (const float*)d_in[0];
    const int*   idx = (const int*)  d_in[1];
    const float* wts = (const float*)d_in[2];
    const float* w1  = (const float*)d_in[3];
    const float* w2  = (const float*)d_in[4];
    const float* w3  = (const float*)d_in[5];
    float* out = (float*)d_out;

    char* ws = (char*)d_ws;
    size_t off = 0;
    bf16* xb  = (bf16*)(ws + off); off += (size_t)N_TOKENS * D_MODEL * 2;
    bf16* w1t = (bf16*)(ws + off); off += (size_t)N_EXPERTS * D_FF * D_MODEL * 2;
    bf16* w2t = (bf16*)(ws + off); off += (size_t)N_EXPERTS * D_FF * D_MODEL * 2;
    bf16* w3t = (bf16*)(ws + off); off += (size_t)N_EXPERTS * D_MODEL * D_FF * 2;
    bf16* H   = (bf16*)(ws + off); off += (size_t)TOTAL * D_FF * 2;
    int*   row_token  = (int*)  (ws + off); off += (size_t)TOTAL * 4;
    float* row_weight = (float*)(ws + off); off += (size_t)TOTAL * 4;
    int* counts   = (int*)(ws + off); off += 256;
    int* offsets  = (int*)(ws + off); off += 256;
    int* counters = (int*)(ws + off); off += 256;
    int* cnts     = (int*)(ws + off); off += 256;

    const int n4 = N_TOKENS * D_MODEL / 4;
    k_init<<<(n4 + 255) / 256, 256, 0, stream>>>(out, n4, counts);
    k_cvt_x<<<(n4 + 255) / 256, 256, 0, stream>>>(x, xb, n4);

    dim3 tb(32, 8);
    k_transpose_bf16<<<dim3(D_FF / 32, D_MODEL / 32, N_EXPERTS), tb, 0, stream>>>(w1, w1t, D_MODEL, D_FF);
    k_transpose_bf16<<<dim3(D_FF / 32, D_MODEL / 32, N_EXPERTS), tb, 0, stream>>>(w2, w2t, D_MODEL, D_FF);
    k_transpose_bf16<<<dim3(D_MODEL / 32, D_FF / 32, N_EXPERTS), tb, 0, stream>>>(w3, w3t, D_FF, D_MODEL);

    k_route_count <<<(TOTAL + 255) / 256, 256, 0, stream>>>(idx, counts);
    k_prefix      <<<1, 1, 0, stream>>>(counts, offsets, counters, cnts);
    k_route_assign<<<(TOTAL + 255) / 256, 256, 0, stream>>>(idx, wts, counters, offsets, cnts,
                                                            row_token, row_weight);

    k_ffn1<<<dim3(D_FF / 128, CAP / 128, N_EXPERTS), 512, 0, stream>>>(
        xb, w1t, w2t, row_token, offsets, cnts, H);
    k_ffn2<<<dim3(D_MODEL / 128, CAP / 128, N_EXPERTS), 512, 0, stream>>>(
        H, w3t, row_token, row_weight, offsets, cnts, out);
}
